// GroupedQueryAttention_48885317763165
// MI455X (gfx1250) — compile-verified, matrix operand order checked by reading, not run on hardware
//
#include <hip/hip_runtime.h>

// ---------------------------------------------------------------------------
// GQA block for MI455X (gfx1250, wave32, WMMA bf16 16x16x32).
// B=2, S=2048, DIM=2048, H=16, KVH=4, HD=128, GROUPS=4, full-head RoPE,
// NON-causal softmax (reference has no mask).
// Round 3: k8-packed B operands (all LDS fragment reads become b128),
// double-buffered async global->LDS staging with partial s_wait_asynccnt.
// ---------------------------------------------------------------------------

typedef __attribute__((ext_vector_type(16))) __bf16 bf16x16;
typedef __attribute__((ext_vector_type(8)))  float  f32x8;

#define S_LEN   2048
#define DIM_    2048
#define NHEADS  16
#define NKV     4
#define HD      128
#define BATCH   2
#define GROUPS_ (NHEADS / NKV)

// ---- async global->LDS copy (CDNA5 path, tracked by ASYNCcnt) -------------
__device__ __forceinline__ void async_copy_b128(const void* gptr, void* lptr) {
    unsigned lds = (unsigned)(unsigned long long)lptr;  // low 32b = LDS offset
    asm volatile("global_load_async_to_lds_b128 %0, %1, off"
                 :: "v"(lds), "v"(gptr) : "memory");
}
template <int N>
__device__ __forceinline__ void wait_async_le() {
    asm volatile("s_wait_asynccnt %0" :: "n"(N) : "memory");
}

// ---- WMMA fragment helpers (layouts per CDNA5 ISA §7.12.2) -----

// A fragment: 16x32 bf16, p -> element (0,0), row-major stride ld.
// Per-lane reads are two contiguous 16B runs -> ds_load_b128 pairs.
__device__ __forceinline__ bf16x16 load_a_frag(const __bf16* p, int ld) {
    const int lane = threadIdx.x & 31;
    const int m    = lane & 15;
    const int kb   = (lane >> 4) << 3;   // 0 or 8
    const __bf16* r = p + (size_t)m * ld;
    bf16x16 a;
#pragma unroll
    for (int h = 0; h < 8; ++h) a[h]     = r[kb + h];
#pragma unroll
    for (int h = 0; h < 8; ++h) a[8 + h] = r[kb + 16 + h];
    return a;
}

// B fragment from k8-packed tile: tile layout [kk8][n][8], kk8 = k/8 (4 groups
// for K=32), ld8 = n_count*8. Element (k,n) at tile[(k>>3)*ld8 + n*8 + (k&7)].
// Per-lane reads: two contiguous 16B runs -> ds_load_b128 pairs.
__device__ __forceinline__ bf16x16 load_b_frag_k8(const __bf16* tile, int nBase, int ld8) {
    const int lane = threadIdx.x & 31;
    const int n    = nBase + (lane & 15);
    const int kk   = lane >> 4;          // kb/8: 0 or 1
    const __bf16* p0 = tile + (size_t)kk * ld8 + n * 8;        // k = kb..kb+7
    const __bf16* p1 = tile + (size_t)(kk + 2) * ld8 + n * 8;  // k = kb+16..kb+23
    bf16x16 b;
#pragma unroll
    for (int h = 0; h < 8; ++h) b[h]     = p0[h];
#pragma unroll
    for (int h = 0; h < 8; ++h) b[8 + h] = p1[h];
    return b;
}

__device__ __forceinline__ f32x8 wmma_bf16(bf16x16 a, bf16x16 b, f32x8 c) {
    return __builtin_amdgcn_wmma_f32_16x16x32_bf16(
        false, a, false, b, (short)0, c, false, false);
}

// ---------------------------------------------------------------------------
// f32 -> bf16 convert (for x, the A operand)
// ---------------------------------------------------------------------------
__global__ void __launch_bounds__(256)
cvt_f32_bf16(const float* __restrict__ src, __bf16* __restrict__ dst, int n) {
    int i = blockIdx.x * 256 + threadIdx.x;
    if (i < n) dst[i] = (__bf16)src[i];
}

// ---------------------------------------------------------------------------
// Weight repack: f32 [K][N] row-major -> bf16 k8-packed [K/8][N][8].
// One thread per (k8, n): 8 coalesced strided reads, one 16B store.
// ---------------------------------------------------------------------------
__global__ void __launch_bounds__(256)
repack_w_k8(const float* __restrict__ w, __bf16* __restrict__ pw, int K, int N) {
    int idx = blockIdx.x * 256 + threadIdx.x;
    if (idx >= (K / 8) * N) return;
    int k8 = idx / N, n = idx % N;
    __bf16* d = pw + (size_t)idx * 8;
#pragma unroll
    for (int j = 0; j < 8; ++j) d[j] = (__bf16)w[(size_t)(k8 * 8 + j) * N + n];
}

// ---------------------------------------------------------------------------
// V repack: vb bf16 [B*S][NKV*HD] -> pv [(b*NKV+kvh)][S/8][HD][8]
// (per kv-head matrix k8-packed over the sequence dimension).
// ---------------------------------------------------------------------------
__global__ void __launch_bounds__(256)
repack_v_k8(const __bf16* __restrict__ v, __bf16* __restrict__ pv) {
    int idx = blockIdx.x * 256 + threadIdx.x;
    const int total = BATCH * NKV * (S_LEN / 8) * HD;
    if (idx >= total) return;
    int d   = idx & (HD - 1);
    int s8  = (idx >> 7) & (S_LEN / 8 - 1);
    int bk  = idx >> 7 >> 8;              // b*NKV + kvh
    int b   = bk / NKV, kvh = bk % NKV;
    __bf16* dst = pv + (size_t)idx * 8;
#pragma unroll
    for (int j = 0; j < 8; ++j)
        dst[j] = v[((size_t)b * S_LEN + s8 * 8 + j) * (NKV * HD) + kvh * HD + d];
}

// ---------------------------------------------------------------------------
// Tiled bf16 WMMA GEMM: C[M,N] = A[M,K] @ B[K,N].
// A bf16 row-major; B bf16 k8-packed [K/8][N][8].
// Block 256 threads (8 waves), tile 256x128, K-step 32, double-buffered
// async staging. Wave w: rows [w*32, w*32+32) x 128 cols = 16 WMMAs/step.
// ---------------------------------------------------------------------------
template <bool OUT_BF16>
__global__ void __launch_bounds__(256)
gemm_bf16(const __bf16* __restrict__ A, const __bf16* __restrict__ PB,
          void* __restrict__ Cv, int M, int N, int K) {
    constexpr int A_TILE = 256 * 32;        // halves per A buffer (16 KB)
    constexpr int B_TILE = 4 * 128 * 8;     // halves per B buffer ( 8 KB)
    __shared__ alignas(16) __bf16 As[2 * A_TILE];
    __shared__ alignas(16) __bf16 Bs[2 * B_TILE];

    const int tid  = threadIdx.x;
    const int wave = tid >> 5;
    const int n0   = blockIdx.x * 128;
    const int m0   = blockIdx.y * 256;
    const int T    = K / 32;

    auto stage = [&](int t, int buf) {
        const int k0 = t * 32;
        // A tile 256x32: 1024 x 16B chunks
        for (int u = tid; u < 1024; u += 256) {
            int row = u >> 2, c4 = u & 3;
            async_copy_b128(A + (size_t)(m0 + row) * K + k0 + c4 * 8,
                            As + buf * A_TILE + row * 32 + c4 * 8);
        }
        // B tile (4 k8-groups x 128 n x 8): 512 x 16B chunks, contiguous rows
        for (int u = tid; u < 512; u += 256) {
            int kk8 = u >> 7, n = u & 127;
            async_copy_b128(PB + ((size_t)((k0 >> 3) + kk8) * N + n0 + n) * 8,
                            Bs + buf * B_TILE + kk8 * (128 * 8) + n * 8);
        }
    };

    f32x8 acc[2][8];
#pragma unroll
    for (int p = 0; p < 2; ++p)
#pragma unroll
        for (int i = 0; i < 8; ++i) acc[p][i] = f32x8{0, 0, 0, 0, 0, 0, 0, 0};

    stage(0, 0);
    for (int t = 0; t < T; ++t) {
        const int cur = t & 1, nxt = cur ^ 1;
        __syncthreads();                     // buf[nxt] consumers (iter t-1) done
        if (t + 1 < T) { stage(t + 1, nxt); wait_async_le<6>(); }
        else          { wait_async_le<0>(); }
        __syncthreads();                     // everyone's tile-t copies resident

        bf16x16 a0 = load_a_frag(As + cur * A_TILE + (wave * 32) * 32, 32);
        bf16x16 a1 = load_a_frag(As + cur * A_TILE + (wave * 32 + 16) * 32, 32);
#pragma unroll
        for (int nt = 0; nt < 8; ++nt) {
            bf16x16 b = load_b_frag_k8(Bs + cur * B_TILE, nt * 16, 128 * 8);
            acc[0][nt] = wmma_bf16(a0, b, acc[0][nt]);
            acc[1][nt] = wmma_bf16(a1, b, acc[1][nt]);
        }
    }

    const int lane = tid & 31;
    const int n    = lane & 15;
    const int hi   = lane >> 4;
#pragma unroll
    for (int p = 0; p < 2; ++p)
#pragma unroll
        for (int nt = 0; nt < 8; ++nt)
#pragma unroll
            for (int i = 0; i < 8; ++i) {
                int m  = m0 + wave * 32 + p * 16 + i + 8 * hi;
                int nn = n0 + nt * 16 + n;
                float v = acc[p][nt][i];
                if (OUT_BF16) ((__bf16*)Cv)[(size_t)m * N + nn] = (__bf16)v;
                else          ((float*)Cv)[(size_t)m * N + nn]  = v;
            }
}

// ---------------------------------------------------------------------------
// RoPE in place on bf16 tensor laid out [B*S, nheads*HD]; full head rotated.
// ---------------------------------------------------------------------------
__global__ void __launch_bounds__(256)
rope_kernel(__bf16* __restrict__ t, int nheads) {
    int idx   = blockIdx.x * 256 + threadIdx.x;
    int total = BATCH * S_LEN * nheads * 64;
    if (idx >= total) return;
    int i   = idx & 63;
    int hh  = (idx >> 6) % nheads;
    int row = idx / (64 * nheads);
    int pos = row % S_LEN;
    float freq = __powf(10000.0f, -(float)(2 * i) * (1.0f / 128.0f));
    float ang  = (float)pos * freq;
    float c = __cosf(ang), s = __sinf(ang);
    __bf16* p = t + (size_t)row * nheads * HD + hh * HD;
    float x1 = (float)p[i];
    float x2 = (float)p[i + 64];
    p[i]      = (__bf16)(x1 * c - x2 * s);
    p[i + 64] = (__bf16)(x1 * s + x2 * c);
}

// ---------------------------------------------------------------------------
// Flash attention (non-causal), GQA. Block = (b, head h, 128 q-rows), 8 waves.
// K streamed row-major (A-style frags), V streamed k8-packed (b128 B frags),
// both double-buffered via async copies. Online softmax in registers; P goes
// C-layout -> A-fragment layout through a 1KB per-wave LDS buffer.
// ---------------------------------------------------------------------------
__global__ void __launch_bounds__(256)
attn_kernel(const __bf16* __restrict__ Q, const __bf16* __restrict__ Kk,
            const __bf16* __restrict__ PV, __bf16* __restrict__ O) {
    constexpr int KV_TILE = 32 * 128;       // 4096 halves (8 KB) per buffer
    __shared__ alignas(16) __bf16 Ks[2 * KV_TILE];
    __shared__ alignas(16) __bf16 Vs[2 * KV_TILE];
    __shared__ alignas(16) __bf16 Pb[8][16 * 32];

    const int tid  = threadIdx.x;
    const int wave = tid >> 5;
    const int lane = tid & 31;
    const int qt   = blockIdx.x;
    const int h    = blockIdx.y;
    const int b    = blockIdx.z;
    const int kvh  = h / GROUPS_;

    const int    qr0      = qt * 128 + wave * 16;
    const size_t qrowBase = (size_t)b * S_LEN + qr0;
    const size_t pvBase   = (size_t)(b * NKV + kvh) * (S_LEN / 8) * (HD * 8);

    auto stage = [&](int kt, int buf) {
        const size_t krow = (size_t)b * S_LEN + kt * 32;
        const __bf16* kp = Kk + krow * (NKV * HD) + kvh * HD;
        for (int u = tid; u < 512; u += 256) {          // K: 32x128 row-major
            int row = u >> 4, c4 = u & 15;
            async_copy_b128(kp + (size_t)row * (NKV * HD) + c4 * 8,
                            Ks + buf * KV_TILE + row * 128 + c4 * 8);
        }
        for (int u = tid; u < 512; u += 256) {          // V: 4 k8-groups x 128 d
            int kk8 = u >> 7, d = u & 127;
            async_copy_b128(PV + pvBase + (size_t)(kt * 4 + kk8) * (HD * 8) + d * 8,
                            Vs + buf * KV_TILE + kk8 * (HD * 8) + d * 8);
        }
    };

    // Q fragments (16 rows x 128 d) in registers
    bf16x16 aq[4];
    {
        const __bf16* qp = Q + qrowBase * (NHEADS * HD) + h * HD;
#pragma unroll
        for (int dc = 0; dc < 4; ++dc) aq[dc] = load_a_frag(qp + dc * 32, NHEADS * HD);
    }

    f32x8 o[8];
#pragma unroll
    for (int i = 0; i < 8; ++i) o[i] = f32x8{0, 0, 0, 0, 0, 0, 0, 0};
    float mi[8], li[8];
#pragma unroll
    for (int i = 0; i < 8; ++i) { mi[i] = -1e30f; li[i] = 0.0f; }

    const float scale = 0.08838834764831845f;  // 1/sqrt(128)
    const int   T     = S_LEN / 32;

    stage(0, 0);
    for (int kt = 0; kt < T; ++kt) {
        const int cur = kt & 1, nxt = cur ^ 1;
        __syncthreads();
        if (kt + 1 < T) { stage(kt + 1, nxt); wait_async_le<4>(); }
        else           { wait_async_le<0>(); }
        __syncthreads();

        const __bf16* ks = Ks + cur * KV_TILE;
        const __bf16* vs = Vs + cur * KV_TILE;

        // scores: S[16q x 32keys] = Q(16x128) @ K^T(128x32)
        f32x8 s0 = f32x8{0, 0, 0, 0, 0, 0, 0, 0};
        f32x8 s1 = f32x8{0, 0, 0, 0, 0, 0, 0, 0};
#pragma unroll
        for (int dc = 0; dc < 4; ++dc) {
            bf16x16 b0 = load_a_frag(ks + dc * 32, 128);
            s0 = wmma_bf16(aq[dc], b0, s0);
            bf16x16 b1 = load_a_frag(ks + 16 * 128 + dc * 32, 128);
            s1 = wmma_bf16(aq[dc], b1, s1);
        }

        // online softmax
        float corr[8];
#pragma unroll
        for (int i = 0; i < 8; ++i) {
            float a0 = s0[i] * scale, a1 = s1[i] * scale;
            float mx = fmaxf(a0, a1);
#pragma unroll
            for (int t = 1; t < 16; t <<= 1) mx = fmaxf(mx, __shfl_xor(mx, t, 32));
            float mnew = fmaxf(mi[i], mx);
            float c    = __expf(mi[i] - mnew);
            float p0   = __expf(a0 - mnew);
            float p1   = __expf(a1 - mnew);
            float rs   = p0 + p1;
#pragma unroll
            for (int t = 1; t < 16; t <<= 1) rs += __shfl_xor(rs, t, 32);
            li[i]   = li[i] * c + rs;
            mi[i]   = mnew;
            corr[i] = c;
            int r = i + 8 * (lane >> 4);
            Pb[wave][r * 32 + (lane & 15)]      = (__bf16)p0;
            Pb[wave][r * 32 + 16 + (lane & 15)] = (__bf16)p1;
        }
#pragma unroll
        for (int nt = 0; nt < 8; ++nt)
#pragma unroll
            for (int i = 0; i < 8; ++i) o[nt][i] *= corr[i];

        // O += P(16x32) @ V(32x128); V fragments are b128 reads (k8-packed)
        bf16x16 pa = load_a_frag(Pb[wave], 32);
#pragma unroll
        for (int nt = 0; nt < 8; ++nt) {
            bf16x16 bv = load_b_frag_k8(vs, nt * 16, HD * 8);
            o[nt] = wmma_bf16(pa, bv, o[nt]);
        }
    }

    // normalize and write attn output as bf16 [B*S, 16*128]
    const int n  = lane & 15;
    const int hi = lane >> 4;
#pragma unroll
    for (int i = 0; i < 8; ++i) {
        float  inv = 1.0f / li[i];
        size_t row = qrowBase + i + 8 * hi;
#pragma unroll
        for (int nt = 0; nt < 8; ++nt)
            O[row * (NHEADS * HD) + h * HD + nt * 16 + n] = (__bf16)(o[nt][i] * inv);
    }
}

// ---------------------------------------------------------------------------
// Host-side launch
// ---------------------------------------------------------------------------
extern "C" void kernel_launch(void* const* d_in, const int* in_sizes, int n_in,
                              void* d_out, int out_size, void* d_ws, size_t ws_size,
                              hipStream_t stream) {
    (void)in_sizes; (void)n_in; (void)out_size; (void)ws_size;
    const float* x  = (const float*)d_in[0];
    const float* wq = (const float*)d_in[1];
    const float* wk = (const float*)d_in[2];
    const float* wv = (const float*)d_in[3];
    const float* wo = (const float*)d_in[4];

    const int MS = BATCH * S_LEN;          // 4096 rows
    char* ws = (char*)d_ws;
    size_t off = 0;
    auto alloc = [&](size_t bytes) { void* p = ws + off; off += (bytes + 255) & ~size_t(255); return p; };

    __bf16* xb    = (__bf16*)alloc((size_t)MS * DIM_ * 2);           // A: x
    __bf16* wqb   = (__bf16*)alloc((size_t)DIM_ * DIM_ * 2);         // k8-packed
    __bf16* wkb   = (__bf16*)alloc((size_t)DIM_ * NKV * HD * 2);     // k8-packed
    __bf16* wvb   = (__bf16*)alloc((size_t)DIM_ * NKV * HD * 2);     // k8-packed
    __bf16* wob   = (__bf16*)alloc((size_t)DIM_ * DIM_ * 2);         // k8-packed
    __bf16* qb    = (__bf16*)alloc((size_t)MS * NHEADS * HD * 2);
    __bf16* kb    = (__bf16*)alloc((size_t)MS * NKV * HD * 2);
    __bf16* vb    = (__bf16*)alloc((size_t)MS * NKV * HD * 2);
    __bf16* pvb   = (__bf16*)alloc((size_t)MS * NKV * HD * 2);       // V k8-packed
    __bf16* attnb = (__bf16*)alloc((size_t)MS * NHEADS * HD * 2);

    cvt_f32_bf16<<<(MS * DIM_ + 255) / 256, 256, 0, stream>>>(x, xb, MS * DIM_);
    auto rpw = [&](const float* s, __bf16* d, int K, int N) {
        int tot = (K / 8) * N;
        repack_w_k8<<<(tot + 255) / 256, 256, 0, stream>>>(s, d, K, N);
    };
    rpw(wq, wqb, DIM_, DIM_);
    rpw(wk, wkb, DIM_, NKV * HD);
    rpw(wv, wvb, DIM_, NKV * HD);
    rpw(wo, wob, DIM_, DIM_);

    dim3 blk(256);
    gemm_bf16<true><<<dim3(DIM_ / 128,       MS / 256), blk, 0, stream>>>(xb, wqb, qb, MS, DIM_,     DIM_);
    gemm_bf16<true><<<dim3((NKV * HD) / 128, MS / 256), blk, 0, stream>>>(xb, wkb, kb, MS, NKV * HD, DIM_);
    gemm_bf16<true><<<dim3((NKV * HD) / 128, MS / 256), blk, 0, stream>>>(xb, wvb, vb, MS, NKV * HD, DIM_);

    {   // RoPE in place on q and k
        int nq = BATCH * S_LEN * NHEADS * 64;
        rope_kernel<<<(nq + 255) / 256, 256, 0, stream>>>(qb, NHEADS);
        int nk = BATCH * S_LEN * NKV * 64;
        rope_kernel<<<(nk + 255) / 256, 256, 0, stream>>>(kb, NKV);
    }

    {   // V k8 repack for the attention B operand
        int tot = BATCH * NKV * (S_LEN / 8) * HD;
        repack_v_k8<<<(tot + 255) / 256, 256, 0, stream>>>(vb, pvb);
    }

    attn_kernel<<<dim3(S_LEN / 128, NHEADS, BATCH), blk, 0, stream>>>(qb, kb, pvb, attnb);

    gemm_bf16<false><<<dim3(DIM_ / 128, MS / 256), blk, 0, stream>>>(attnb, wob, d_out, MS, DIM_, DIM_);
}
